// EfficientViMBlock_61804579390012
// MI455X (gfx1250) — matrix operand order repaired and checked by reading, not compiled, not run on hardware
//
#include <hip/hip_runtime.h>
#include <hip/hip_bf16.h>

// EfficientViM block for MI455X (gfx1250, wave32).
// All 7 GEMMs go through v_wmma_f32_16x16x32_f16 with f32 accumulation.
// Workspace requirement: 50331648 floats = 201,326,592 bytes.

typedef _Float16 v16h __attribute__((ext_vector_type(16)));
typedef _Float16 v8h  __attribute__((ext_vector_type(8)));
typedef _Float16 v4h  __attribute__((ext_vector_type(4)));
typedef float    v8f  __attribute__((ext_vector_type(8)));
typedef float    v4f  __attribute__((ext_vector_type(4)));

#define BMT 128
#define BNT 128
#define BKT 32
#define LDT 40   // padded LDS row stride in halfs (80 B: multiple of 16 B)

// ---------------------------------------------------------------------------
// Fragment loader: two 16-byte LDS reads -> one 16-half WMMA operand.
// ---------------------------------------------------------------------------
__device__ __forceinline__ v16h frag2x8(const _Float16* p0, const _Float16* p1) {
    v8h a = *(const v8h*)p0;
    v8h b = *(const v8h*)p1;
    return __builtin_shufflevector(a, b, 0,1,2,3,4,5,6,7,8,9,10,11,12,13,14,15);
}

// ---------------------------------------------------------------------------
// Tiled WMMA GEMM: C[M,N](f32 or f16) = A[M,K](f32) * B[K,N](f32 or f16)
// 256 threads = 8 waves in a 2(M) x 4(N) grid; each wave computes 64x32 via
// 4x2 fragments of v_wmma_f32_16x16x32_f16. K must be a multiple of 32;
// m0/n0 tile origins are multiples of 128, so all vector loads are 16B-aligned.
// Staging is branch-free: OOB rows/cols load a clamped valid address and are
// zeroed with selects, keeping EXEC full and letting loads batch (1 wait).
// strideA/strideB/strideC are per-batch element strides (blockIdx.z = batch).
// ---------------------------------------------------------------------------
template<bool BHALF, bool OUTHALF>
__global__ __launch_bounds__(256) void gemm_wmma_kernel(
    const float* __restrict__ A, size_t strideA,
    const void*  __restrict__ B, size_t strideB,
    void*        __restrict__ C, size_t strideC,
    int M, int N, int K)
{
    __shared__ __align__(16) _Float16 As[BMT * LDT];
    __shared__ __align__(16) _Float16 Bs[BNT * LDT];

    const int bz  = blockIdx.z;
    const float* Ab = A + (size_t)bz * strideA;
    const int m0 = blockIdx.x * BMT;
    const int n0 = blockIdx.y * BNT;
    const int tid  = threadIdx.x;
    const int lane = tid & 31;
    const int wv   = tid >> 5;
    const int wm   = wv >> 2;      // 0..1
    const int wn   = wv & 3;       // 0..3
    const int h16  = lane & 15;
    const int kA   = (lane >= 16) ? 8  : 0;   // A frag K sub-offset (ISA layout)
    const int kB   = (lane >= 16) ? 8  : 0;   // B frag K sub-offset (ISA layout)

    const v4f zero4 = {0.f, 0.f, 0.f, 0.f};
    v8f acc[4][2] = {};

    for (int k0 = 0; k0 < K; k0 += BKT) {
        // ---- Phase 1: batched, branch-free global loads into registers ----
        v4f aReg[4];
        #pragma unroll
        for (int i = 0; i < 4; ++i) {
            int idx4 = tid + i * 256;        // 1024 float4 = 128 rows x 8
            int r = idx4 >> 3, c4 = idx4 & 7;
            int gm = m0 + r;
            bool ok = gm < M;
            const v4f* pa = (const v4f*)(Ab + (size_t)(ok ? gm : 0) * K + k0) + c4;
            v4f v = *pa;
            aReg[i] = ok ? v : zero4;
        }

        v4f bReg4[4];
        v8h bReg8[2];
        if constexpr (!BHALF) {
            const float* Bf = (const float*)B + (size_t)bz * strideB;
            #pragma unroll
            for (int i = 0; i < 4; ++i) {
                int idx4 = tid + i * 256;    // 1024 float4 = 32 rows x 32
                int kk = idx4 >> 5, n4 = (idx4 & 31) << 2;
                int gn = n0 + n4;
                bool ok = gn < N;
                const v4f* pb = (const v4f*)(Bf + (size_t)(k0 + kk) * N) +
                                ((ok ? gn : 0) >> 2);
                v4f v = *pb;
                bReg4[i] = ok ? v : zero4;
            }
        } else {
            const _Float16* Bh = (const _Float16*)B + (size_t)bz * strideB;
            const v8h zero8 = {};
            #pragma unroll
            for (int i = 0; i < 2; ++i) {
                int idx8 = tid + i * 256;    // 512 v8h = 32 rows x 16
                int kk = idx8 >> 4, n8 = (idx8 & 15) << 3;
                int gn = n0 + n8;
                bool ok = gn < N;
                const v8h* pb = (const v8h*)(Bh + (size_t)(k0 + kk) * N) +
                                ((ok ? gn : 0) >> 3);
                v8h v = *pb;
                bReg8[i] = ok ? v : zero8;
            }
        }

        // Prefetch next K-tile (global_prefetch_b8); everything is L2-resident.
        if (k0 + BKT < K) {
            int r = tid >> 3, c4 = tid & 7;
            int gm = m0 + r;
            __builtin_prefetch(Ab + (size_t)(gm < M ? gm : 0) * K + (k0 + BKT) + c4 * 4, 0, 1);
        }

        // ---- Phase 2: convert + store to LDS ----
        #pragma unroll
        for (int i = 0; i < 4; ++i) {
            int idx4 = tid + i * 256;
            int r = idx4 >> 3, c4 = idx4 & 7;
            v4h hv = { (_Float16)aReg[i].x, (_Float16)aReg[i].y,
                       (_Float16)aReg[i].z, (_Float16)aReg[i].w };
            *(v4h*)&As[r * LDT + (c4 << 2)] = hv;     // 8B ds_store_b64
        }
        if constexpr (!BHALF) {
            #pragma unroll
            for (int i = 0; i < 4; ++i) {
                int idx4 = tid + i * 256;
                int kk = idx4 >> 5, n4 = (idx4 & 31) << 2;
                Bs[(n4 + 0) * LDT + kk] = (_Float16)bReg4[i].x;  // transpose scatter
                Bs[(n4 + 1) * LDT + kk] = (_Float16)bReg4[i].y;
                Bs[(n4 + 2) * LDT + kk] = (_Float16)bReg4[i].z;
                Bs[(n4 + 3) * LDT + kk] = (_Float16)bReg4[i].w;
            }
        } else {
            #pragma unroll
            for (int i = 0; i < 2; ++i) {
                int idx8 = tid + i * 256;
                int kk = idx8 >> 4, n8 = (idx8 & 15) << 3;
                #pragma unroll
                for (int j = 0; j < 8; ++j)
                    Bs[(n8 + j) * LDT + kk] = bReg8[i][j];
            }
        }
        __syncthreads();

        // ---- Phase 3: fragment loads + 8 WMMAs per wave ----
        v16h af[4], bf[2];
        #pragma unroll
        for (int i = 0; i < 4; ++i) {
            const _Float16* p = &As[(wm * 64 + i * 16 + h16) * LDT];
            af[i] = frag2x8(p + kA, p + 16 + kA);   // K {0..7,16..23} / {8..15,24..31}
        }
        #pragma unroll
        for (int j = 0; j < 2; ++j) {
            const _Float16* p = &Bs[(wn * 32 + j * 16 + h16) * LDT] + kB;
            bf[j] = frag2x8(p, p + 16);             // K {0..7,16..23} / {8..15,24..31}
        }
        #pragma unroll
        for (int i = 0; i < 4; ++i)
            #pragma unroll
            for (int j = 0; j < 2; ++j)
                acc[i][j] = __builtin_amdgcn_wmma_f32_16x16x32_f16(
                    false, af[i], false, bf[j], (short)0, acc[i][j], false, false);
        __syncthreads();
    }

    // Store per documented C/D layout: VGPR e -> row (e or 8+e), col lane%16.
    const int rofs = (lane >= 16) ? 8 : 0;
    for (int i = 0; i < 4; ++i) {
        int gmBase = m0 + wm * 64 + i * 16 + rofs;
        for (int j = 0; j < 2; ++j) {
            int gn = n0 + wn * 32 + j * 16 + h16;
            if (gn >= N) continue;
            #pragma unroll
            for (int e = 0; e < 8; ++e) {
                int gm = gmBase + e;
                if (gm < M) {
                    size_t off = (size_t)bz * strideC + (size_t)gm * N + gn;
                    if (OUTHALF) ((_Float16*)C)[off] = (_Float16)acc[i][j][e];
                    else         ((float*)C)[off]    = acc[i][j][e];
                }
            }
        }
    }
}

// ---------------------------------------------------------------------------
// Depthwise 3x3 (+ optional BN eval + optional sigmoid(alpha) residual mix).
// One block per (b,c) 32x32 plane staged in LDS.
// mode bit0: BN, bit1: residual alpha mix.
// ---------------------------------------------------------------------------
__global__ __launch_bounds__(256) void dw3x3_kernel(
    const float* __restrict__ x, const float* __restrict__ w,
    const float* __restrict__ bng, const float* __restrict__ bnb,
    const float* __restrict__ bnm, const float* __restrict__ bnv,
    const float* __restrict__ alpha_row, float* __restrict__ out,
    int C, int mode)
{
    __shared__ float tile[1024];
    const int bc = blockIdx.x;
    const int c  = bc % C;
    const float* p = x + (size_t)bc * 1024;
    const int tid = threadIdx.x;
    #pragma unroll
    for (int i = 0; i < 4; ++i) tile[tid + i * 256] = p[tid + i * 256];
    __syncthreads();

    float wv[9];
    #pragma unroll
    for (int i = 0; i < 9; ++i) wv[i] = w[c * 9 + i];
    float scale = 1.f, shift = 0.f;
    if (mode & 1) { scale = bng[c] * rsqrtf(bnv[c] + 1e-5f); shift = bnb[c] - bnm[c] * scale; }
    float a = 1.f;
    if (mode & 2) a = 1.f / (1.f + expf(-alpha_row[c]));

    #pragma unroll
    for (int i = 0; i < 4; ++i) {
        int idx = tid + i * 256;
        int py = idx >> 5, px = idx & 31;
        float s = 0.f;
        for (int dy = -1; dy <= 1; ++dy) {
            int yy = py + dy; if (yy < 0 || yy > 31) continue;
            for (int dx = -1; dx <= 1; ++dx) {
                int xx = px + dx; if (xx < 0 || xx > 31) continue;
                s += tile[yy * 32 + xx] * wv[(dy + 1) * 3 + (dx + 1)];
            }
        }
        if (mode & 1) s = s * scale + shift;
        float o = (mode & 2) ? ((1.f - a) * tile[idx] + a * s) : s;
        out[(size_t)bc * 1024 + idx] = o;
    }
}

// ---------------------------------------------------------------------------
// Channel LayerNorm (C=256) per (b,l) position; one block per position.
// ---------------------------------------------------------------------------
__global__ __launch_bounds__(256) void ln_chan_kernel(
    const float* __restrict__ xin, const float* __restrict__ lw,
    const float* __restrict__ lb, float* __restrict__ xout)
{
    __shared__ float rs[256], rq[256];
    const int pos = blockIdx.x;                 // b*1024 + l
    const int b = pos >> 10, l = pos & 1023;
    const int c = threadIdx.x;
    const size_t base = ((size_t)b * 256 + c) * 1024 + l;
    float v = xin[base];
    rs[c] = v; rq[c] = v * v;
    __syncthreads();
    for (int o = 128; o > 0; o >>= 1) {
        if (c < o) { rs[c] += rs[c + o]; rq[c] += rq[c + o]; }
        __syncthreads();
    }
    float mean = rs[0] * (1.f / 256.f);
    float var  = rq[0] * (1.f / 256.f) - mean * mean;
    float inv  = rsqrtf(var + 1e-6f);
    xout[base] = lw[c] * ((v - mean) * inv) + lb[c];
}

// ---------------------------------------------------------------------------
// Softmax over L of (dt + A[s]) and ABt[l][s] = Am*Bm (transposed write so the
// h-einsum becomes a row-major GEMM). One block per (b,s).
// bcdt2 layout [b,192,1024]: rows 0..63 = Bm, 64..127 = Cm, 128..191 = dt.
// ---------------------------------------------------------------------------
__global__ __launch_bounds__(256) void softmax_ab_kernel(
    const float* __restrict__ bcdt2, const float* __restrict__ Avec,
    float* __restrict__ ABt)
{
    __shared__ float red[256];
    const int b = blockIdx.x >> 6, s = blockIdx.x & 63;
    const float* dt = bcdt2 + ((size_t)b * 192 + 128 + s) * 1024;
    const float* Bm = bcdt2 + ((size_t)b * 192 + s) * 1024;
    const float Av = Avec[s];
    const int tid = threadIdx.x;

    float lv[4];
    float mx = -3.4e38f;
    #pragma unroll
    for (int i = 0; i < 4; ++i) { lv[i] = dt[tid + i * 256] + Av; mx = fmaxf(mx, lv[i]); }
    red[tid] = mx; __syncthreads();
    for (int o = 128; o > 0; o >>= 1) {
        if (tid < o) red[tid] = fmaxf(red[tid], red[tid + o]);
        __syncthreads();
    }
    mx = red[0]; __syncthreads();

    float sm = 0.f;
    #pragma unroll
    for (int i = 0; i < 4; ++i) { lv[i] = expf(lv[i] - mx); sm += lv[i]; }
    red[tid] = sm; __syncthreads();
    for (int o = 128; o > 0; o >>= 1) {
        if (tid < o) red[tid] += red[tid + o];
        __syncthreads();
    }
    float inv = 1.f / red[0];
    #pragma unroll
    for (int i = 0; i < 4; ++i) {
        int l = tid + i * 256;
        ABt[((size_t)b * 1024 + l) * 64 + s] = lv[i] * inv * Bm[l];
    }
}

// ---------------------------------------------------------------------------
// hmix = hh * silu(z) + hh * D  (hz layout [b,512,64]; hh rows 0..255, z 256..511)
// ---------------------------------------------------------------------------
__global__ void silu_mix_kernel(const float* __restrict__ hz,
                                const float* __restrict__ D,
                                float* __restrict__ hmix, int total)
{
    int idx = blockIdx.x * 256 + threadIdx.x;
    if (idx >= total) return;
    int b = idx / (256 * 64);
    int rem = idx - b * 256 * 64;
    int c = rem >> 6, s = rem & 63;
    size_t base = (size_t)b * 512 * 64;
    float hh = hz[base + (size_t)c * 64 + s];
    float z  = hz[base + (size_t)(256 + c) * 64 + s];
    float sz = z / (1.f + expf(-z));
    hmix[idx] = hh * sz + hh * D[0];
}

// ---------------------------------------------------------------------------
// out = (1-sigmoid(alpha[c]))*xa + sigmoid(alpha[c])*xb  over [B,256,1024]
// ---------------------------------------------------------------------------
__global__ void mix_kernel(const float* __restrict__ xa, const float* __restrict__ xb,
                           const float* __restrict__ alpha_row,
                           float* __restrict__ out, int total)
{
    int idx = blockIdx.x * 256 + threadIdx.x;
    if (idx >= total) return;
    int c = (idx >> 10) & 255;
    float a = 1.f / (1.f + expf(-alpha_row[c]));
    out[idx] = (1.f - a) * xa[idx] + a * xb[idx];
}

// ---------------------------------------------------------------------------
// In-place LayerNorm (C=1024) + exact GELU on the f16 FFN hidden tensor.
// One block per (b,l) position; 4 channels per thread.
// ---------------------------------------------------------------------------
__global__ __launch_bounds__(256) void ln_gelu_f16_kernel(
    _Float16* __restrict__ f, const float* __restrict__ lw,
    const float* __restrict__ lb)
{
    __shared__ float rs[256], rq[256];
    const int pos = blockIdx.x;
    const int b = pos >> 10, l = pos & 1023;
    const int tid = threadIdx.x;

    float v[4], s = 0.f, q = 0.f;
    #pragma unroll
    for (int i = 0; i < 4; ++i) {
        int ch = tid + i * 256;
        v[i] = (float)f[((size_t)b * 1024 + ch) * 1024 + l];
        s += v[i]; q += v[i] * v[i];
    }
    rs[tid] = s; rq[tid] = q; __syncthreads();
    for (int o = 128; o > 0; o >>= 1) {
        if (tid < o) { rs[tid] += rs[tid + o]; rq[tid] += rq[tid + o]; }
        __syncthreads();
    }
    float mean = rs[0] * (1.f / 1024.f);
    float var  = rq[0] * (1.f / 1024.f) - mean * mean;
    float inv  = rsqrtf(var + 1e-6f);
    #pragma unroll
    for (int i = 0; i < 4; ++i) {
        int ch = tid + i * 256;
        float t = lw[ch] * ((v[i] - mean) * inv) + lb[ch];
        t = 0.5f * t * (1.f + erff(t * 0.70710678118654752f));
        f[((size_t)b * 1024 + ch) * 1024 + l] = (_Float16)t;
    }
}

// ---------------------------------------------------------------------------
extern "C" void kernel_launch(void* const* d_in, const int* in_sizes, int n_in,
                              void* d_out, int out_size, void* d_ws, size_t ws_size,
                              hipStream_t stream)
{
    (void)in_sizes; (void)n_in; (void)out_size; (void)ws_size;

    const float* x      = (const float*)d_in[0];
    const float* alpha  = (const float*)d_in[1];
    const float* dw1_w  = (const float*)d_in[2];
    const float* bn1_g  = (const float*)d_in[3];
    const float* bn1_b  = (const float*)d_in[4];
    const float* bn1_m  = (const float*)d_in[5];
    const float* bn1_v  = (const float*)d_in[6];
    const float* dw2_w  = (const float*)d_in[7];
    const float* bn2_g  = (const float*)d_in[8];
    const float* bn2_b  = (const float*)d_in[9];
    const float* bn2_m  = (const float*)d_in[10];
    const float* bn2_v  = (const float*)d_in[11];
    const float* ln_w   = (const float*)d_in[12];
    const float* ln_b   = (const float*)d_in[13];
    const float* bcdt_w = (const float*)d_in[14];
    const float* dwm_w  = (const float*)d_in[15];
    const float* hz_w   = (const float*)d_in[16];
    const float* outp_w = (const float*)d_in[17];
    const float* Avec   = (const float*)d_in[18];
    const float* Dp     = (const float*)d_in[19];
    const float* fc1_w  = (const float*)d_in[20];
    const float* fc1lw  = (const float*)d_in[21];
    const float* fc1lb  = (const float*)d_in[22];
    const float* fc2_w  = (const float*)d_in[23];

    // Workspace layout (element offsets in floats); total 50331648 f32 = 192 MiB.
    float* ws    = (float*)d_ws;
    float* x1    = ws;                    //  8388608  [B,256,1024]  (later reused for fc2 out)
    float* xn    = ws + 8388608;          //  8388608  [B,256,1024]
    float* bcdt  = ws + 16777216;         //  6291456  [B,192,1024]
    float* ABt   = ws + 23068672;         //  2097152  [B,1024,64]
    float* bcdt2 = ws + 25165824;         //  6291456  [B,192,1024]
    float* h     = ws + 31457280;         //   524288  [B,256,64]
    float* hz    = ws + 31981568;         //  1048576  [B,512,64]
    float* hmix  = ws + 33030144;         //   524288  [B,256,64]
    float* y     = ws + 33554432;         //  8388608  [B,256,1024]  (x2 in-place)
    float* x3    = ws + 41943040;         //  8388608  [B,256,1024]
    // f16 FFN hidden [B,1024,1024] overlays xn+bcdt+ABt (dead by then): exactly 64 MiB.
    _Float16* f  = (_Float16*)(ws + 8388608);
    float* fc2o  = x1;                    // x1 dead after mix1

    float* out_x    = (float*)d_out;              // [B,256,32,32]
    float* out_hout = (float*)d_out + 8388608;    // [B,256,64]

    const dim3 blk(256);

    // 1. dwconv1 + BN + alpha0 mix
    dw3x3_kernel<<<dim3(32 * 256), blk, 0, stream>>>(
        x, dw1_w, bn1_g, bn1_b, bn1_m, bn1_v, alpha + 0 * 256, x1, 256, 3);
    // 2. channel LayerNorm
    ln_chan_kernel<<<dim3(32768), blk, 0, stream>>>(x1, ln_w, ln_b, xn);
    // 3. bcdt = bcdt_w[192,256] @ xn[b][256,1024]
    gemm_wmma_kernel<false, false><<<dim3(2, 8, 32), blk, 0, stream>>>(
        bcdt_w, 0, xn, (size_t)256 * 1024, bcdt, (size_t)192 * 1024, 192, 1024, 256);
    // 4. depthwise 3x3 on bcdt
    dw3x3_kernel<<<dim3(32 * 192), blk, 0, stream>>>(
        bcdt, dwm_w, nullptr, nullptr, nullptr, nullptr, nullptr, bcdt2, 192, 0);
    // 5. softmax(dt+A) * Bm -> ABt (transposed)
    softmax_ab_kernel<<<dim3(32 * 64), blk, 0, stream>>>(bcdt2, Avec, ABt);
    // 6. h = xn[b][256,1024] @ ABt[b][1024,64]
    gemm_wmma_kernel<false, false><<<dim3(2, 1, 32), blk, 0, stream>>>(
        xn, (size_t)256 * 1024, ABt, (size_t)1024 * 64, h, (size_t)256 * 64, 256, 64, 1024);
    // 7. hz = hz_w[512,256] @ h[b][256,64]
    gemm_wmma_kernel<false, false><<<dim3(4, 1, 32), blk, 0, stream>>>(
        hz_w, 0, h, (size_t)256 * 64, hz, (size_t)512 * 64, 512, 64, 256);
    // 8. hmix = hh*silu(z) + hh*D
    silu_mix_kernel<<<dim3(2048), blk, 0, stream>>>(hz, Dp, hmix, 32 * 256 * 64);
    // 9. hout = outp_w[256,256] @ hmix[b][256,64]  (written straight to output #2)
    gemm_wmma_kernel<false, false><<<dim3(2, 1, 32), blk, 0, stream>>>(
        outp_w, 0, hmix, (size_t)256 * 64, out_hout, (size_t)256 * 64, 256, 64, 256);
    // 10. y = hout[b][256,64] @ Cm[b][64,1024]  (Cm = bcdt2 rows 64..127)
    gemm_wmma_kernel<false, false><<<dim3(2, 8, 32), blk, 0, stream>>>(
        out_hout, (size_t)256 * 64, bcdt2 + (size_t)64 * 1024, (size_t)192 * 1024,
        y, (size_t)256 * 1024, 256, 1024, 64);
    // 11. x2 = (1-a1)*x1 + a1*y  (in place on y)
    mix_kernel<<<dim3(32768), blk, 0, stream>>>(x1, y, alpha + 1 * 256, y, 8388608);
    // 12. dwconv2 + BN + alpha2 mix
    dw3x3_kernel<<<dim3(32 * 256), blk, 0, stream>>>(
        y, dw2_w, bn2_g, bn2_b, bn2_m, bn2_v, alpha + 2 * 256, x3, 256, 3);
    // 13. f(f16) = fc1_w[1024,256] @ x3[b][256,1024]
    gemm_wmma_kernel<false, true><<<dim3(8, 8, 32), blk, 0, stream>>>(
        fc1_w, 0, x3, (size_t)256 * 1024, f, (size_t)1024 * 1024, 1024, 1024, 256);
    // 14. in-place LayerNorm(1024) + exact GELU on f
    ln_gelu_f16_kernel<<<dim3(32768), blk, 0, stream>>>(f, fc1lw, fc1lb);
    // 15. fc2o = fc2_w[256,1024] @ f[b][1024,1024]  (f16 B operand)
    gemm_wmma_kernel<true, false><<<dim3(2, 8, 32), blk, 0, stream>>>(
        fc2_w, 0, f, (size_t)1024 * 1024, fc2o, (size_t)256 * 1024, 256, 1024, 1024);
    // 16. out = (1-a3)*x3 + a3*fc2o
    mix_kernel<<<dim3(32768), blk, 0, stream>>>(x3, fc2o, alpha + 3 * 256, out_x, 8388608);
}
